// DpEmbeddingNet_7765300871595
// MI455X (gfx1250) — matrix operand order repaired and checked by reading, not compiled, not run on hardware
//
#include <hip/hip_runtime.h>
#include <math.h>

typedef _Float16 f16;
typedef __attribute__((ext_vector_type(16))) _Float16 v16h;
typedef __attribute__((ext_vector_type(8)))  _Float16 v8h;
typedef __attribute__((ext_vector_type(8)))  float    v8f;

#define WPB 8   // waves per block (256 threads, wave32)
#define TPW 8   // 16-row tiles per wave

static __device__ __forceinline__ v8f wmma_f16(v16h a, v16h b, v8f c) {
    // D = A(16x32 f16) * B(32x16 f16) + C(16x16 f32)
    return __builtin_amdgcn_wmma_f32_16x16x32_f16(
        /*neg_a=*/false, a, /*neg_b=*/false, b,
        /*c_mod=*/(short)0, c, /*reuse_a=*/false, /*reuse_b=*/false);
}

// Branchless tanh: minimax odd poly for |x|<0.625, hardware exp2/rcp tail
// (1 - 2/(e^{2|x|}+1), saturates correctly to +-1), merged via cndmask.
// ~15 VALU/TRANS ops, no divergent control flow, TRANS co-executes with VALU.
static __device__ __forceinline__ float fast_tanh(float x) {
    const float u = x * x;
    // tanh(x) = x + x^3 * P(x^2) on [-0.625, 0.625]
    float p = __builtin_fmaf(u, -4.89632212e-3f,  2.06306398e-2f);
    p       = __builtin_fmaf(u, p,               -5.37385419e-2f);
    p       = __builtin_fmaf(u, p,                1.33310467e-1f);
    p       = __builtin_fmaf(u, p,               -3.33332062e-1f);
    const float poly = __builtin_fmaf(x * u, p, x);
    // tail: t = 1 - 2/(exp2(2*log2(e)*|x|) + 1), then copysign
    const float e  = __builtin_amdgcn_exp2f(__builtin_fabsf(x) * 2.88539008e0f);
    const float r  = __builtin_amdgcn_rcpf(e + 1.0f);
    const float t  = __builtin_copysignf(__builtin_fmaf(r, -2.0f, 1.0f), x);
    return (__builtin_fabsf(x) < 0.625f) ? poly : t;
}

__launch_bounds__(256)
__global__ void dp_embed_wmma(const float* __restrict__ xx,
                              const float* __restrict__ w0, const float* __restrict__ b0,
                              const float* __restrict__ r0,
                              const float* __restrict__ w1, const float* __restrict__ b1,
                              const float* __restrict__ r1,
                              const float* __restrict__ w2, const float* __restrict__ b2,
                              const float* __restrict__ r2,
                              float* __restrict__ out, int rows, int nTiles)
{
    // ---- LDS staging (per block) -------------------------------------
    __shared__ __attribute__((aligned(64))) f16 w1hT[64][32];    // B1 transposed: [N][K], zero-padded
    __shared__ __attribute__((aligned(64))) f16 w2hT[112][64];   // B2 transposed: [N][K], zero-padded
    __shared__ float w0f[32], b0f[32];
    __shared__ float b1f[64], r1f[64];
    __shared__ float b2f[112], r2f[112];
    __shared__ float xbuf[WPB][16];
    __shared__ float h0f[WPB][16][26];                            // layer-0 activations (f32 residual source)
    __shared__ __attribute__((aligned(64))) f16 xx1h[WPB][16][64]; // layer-1 output, f16, zero-padded K

    const int tid = threadIdx.x;

    for (int idx = tid; idx < 64 * 32; idx += 256) {
        int n = idx >> 5, k = idx & 31;
        w1hT[n][k] = (k < 25 && n < 50) ? (f16)w1[k * 50 + n] : (f16)0.f;
    }
    for (int idx = tid; idx < 112 * 64; idx += 256) {
        int n = idx >> 6, k = idx & 63;
        w2hT[n][k] = (k < 50 && n < 100) ? (f16)w2[k * 100 + n] : (f16)0.f;
    }
    if (tid < 32)  { w0f[tid] = (tid < 25)  ? w0[tid] : 0.f;
                     b0f[tid] = (tid < 25)  ? b0[tid] : 0.f; }
    if (tid < 64)  { b1f[tid] = (tid < 50)  ? b1[tid] : 0.f;
                     r1f[tid] = (tid < 50)  ? r1[tid] : 0.f; }
    if (tid < 112) { b2f[tid] = (tid < 100) ? b2[tid] : 0.f;
                     r2f[tid] = (tid < 100) ? r2[tid] : 0.f; }
    (void)r0; // layer 0: d_out != d_in and != 2*d_in -> no residual term

    // zero the invariant K-padding columns (n = 50..63) of xx1h once
    {
        const int lane = tid & 31, wv = tid >> 5;
        for (int idx = lane; idx < 16 * 14; idx += 32) {
            int m = idx / 14, n = 50 + idx % 14;
            xx1h[wv][m][n] = (f16)0.f;
        }
    }
    __syncthreads();

    const int lane   = tid & 31;
    const int wv     = tid >> 5;
    const int laneN  = lane & 15;           // column / row-in-tile index
    const int hiHalf = lane >> 4;           // 0 or 1
    const int ka     = hiHalf ? 16 : 0;     // A K base: contiguous order k = 16h + i, same as B
    const int kb     = hiHalf ? 16 : 0;     // B-layout K base

    for (int it = 0; it < TPW; ++it) {
        const int tile    = (blockIdx.x * WPB + wv) * TPW + it;
        const bool active = tile < nTiles;
        const int rowBase = tile * 16;

        // ---- stage x tile (layer-0 input, 1 value per row) ----------
        if (lane < 16) {
            int r = rowBase + lane;
            xbuf[wv][lane] = (active && r < rows) ? xx[r] : 0.f;
            if (active && tile + 1 < nTiles)
                __builtin_prefetch(xx + rowBase + 16 + lane, 0, 0); // global_prefetch_b8
        }
        __syncthreads();

        // ---- layer 0: h0[m][k] = tanh(x[m]*w0[k]+b0[k]), built
        //      directly in the 16x32 f16 A-matrix register layout -----
        const float xv = xbuf[wv][laneN];
        v16h Amat;
#pragma unroll
        for (int i = 0; i < 16; ++i) {
            int K = ka + i; // lanes<16: K in {0-15}; lanes>=16: {16-31}
            float h = 0.f;
            if (K < 25) {
                h = fast_tanh(__builtin_fmaf(xv, w0f[K], b0f[K]));
                h0f[wv][laneN][K] = h;               // each (m,K<25) owned by exactly one lane
            }
            Amat[i] = (f16)h;
        }
        __syncthreads();

        // ---- layer 1: 16x32 @ 32x64 (4 N-tiles), epilogue ->
        //      xx1 = tanh(h0@w1+b1)*r1 + concat(h0,h0) ---------------
#pragma unroll
        for (int t = 0; t < 4; ++t) {
            const int n = 16 * t + laneN;
            v16h Bv = *reinterpret_cast<const v16h*>(&w1hT[n][kb]); // contiguous 32B: K=kb..kb+15
            v8f acc = {};
            acc = wmma_f16(Amat, Bv, acc);
            if (n < 50) {
                const float bb = b1f[n], rr = r1f[n];
                const int   c  = (n < 25) ? n : (n - 25);
#pragma unroll
                for (int v = 0; v < 8; ++v) {
                    int mm = hiHalf ? (v + 8) : v;   // C/D layout: lane group selects M half
                    float a = fast_tanh(acc[v] + bb);
                    xx1h[wv][mm][n] = (f16)__builtin_fmaf(a, rr, h0f[wv][mm][c]);
                }
            }
        }
        __syncthreads();

        // ---- layer 2: 16x64 @ 64x112 (7 N-tiles, 2 K-steps) --------
        union { v16h v; v8h h[2]; } A0, A1;          // A operands for K=0..31 and K=32..63
        A0.h[0] = *reinterpret_cast<const v8h*>(&xx1h[wv][laneN][ka]);
        A0.h[1] = *reinterpret_cast<const v8h*>(&xx1h[wv][laneN][ka + 8]);
        A1.h[0] = *reinterpret_cast<const v8h*>(&xx1h[wv][laneN][32 + ka]);
        A1.h[1] = *reinterpret_cast<const v8h*>(&xx1h[wv][laneN][32 + ka + 8]);

#pragma unroll
        for (int t = 0; t < 7; ++t) {
            const int n = 16 * t + laneN;
            v16h B0 = *reinterpret_cast<const v16h*>(&w2hT[n][kb]);
            v16h B1 = *reinterpret_cast<const v16h*>(&w2hT[n][32 + kb]);
            v8f acc = {};
            acc = wmma_f16(A0.v, B0, acc);
            acc = wmma_f16(A1.v, B1, acc);
            if (active && n < 100) {
                const float bb = b2f[n], rr = r2f[n];
                const int   c  = (n < 50) ? n : (n - 50);
#pragma unroll
                for (int v = 0; v < 8; ++v) {
                    int mm  = hiHalf ? (v + 8) : v;
                    int row = rowBase + mm;
                    if (row < rows) {
                        float a   = fast_tanh(acc[v] + bb);
                        float val = __builtin_fmaf(a, rr, (float)xx1h[wv][mm][c]);
                        // 315 MB streaming output (> 192 MB L2, never re-read):
                        // non-temporal store to avoid cache rinse
                        __builtin_nontemporal_store(val, &out[(long)row * 100 + n]);
                    }
                }
            }
        }
        __syncthreads();
    }
}

extern "C" void kernel_launch(void* const* d_in, const int* in_sizes, int n_in,
                              void* d_out, int out_size, void* d_ws, size_t ws_size,
                              hipStream_t stream) {
    (void)n_in; (void)out_size; (void)d_ws; (void)ws_size;
    const float* xx = (const float*)d_in[0];
    const float* w0 = (const float*)d_in[1];
    const float* b0 = (const float*)d_in[2];
    const float* r0 = (const float*)d_in[3];
    const float* w1 = (const float*)d_in[4];
    const float* b1 = (const float*)d_in[5];
    const float* r1 = (const float*)d_in[6];
    const float* w2 = (const float*)d_in[7];
    const float* b2 = (const float*)d_in[8];
    const float* r2 = (const float*)d_in[9];
    float* out = (float*)d_out;

    const int rows   = in_sizes[0];                  // 32*192*128*1 = 786432
    const int nTiles = (rows + 15) / 16;             // 49152 row-tiles of 16
    const int tilesPerBlock = WPB * TPW;             // 64
    const int blocks = (nTiles + tilesPerBlock - 1) / tilesPerBlock;

    dp_embed_wmma<<<blocks, 256, 0, stream>>>(xx, w0, b0, r0, w1, b1, r1,
                                              w2, b2, r2, out, rows, nTiles);
}
